// ModRNNBase_54726473286107
// MI455X (gfx1250) — matrix operand order repaired and checked by reading, not compiled, not run on hardware
//
#include <hip/hip_runtime.h>

#define T_STEPS 512
#define BATCH   64
#define ISIZE   1024
#define HSIZE   1024
#define GSIZE   4096   // 4*H (i,f,g,o quadrants)
#define KSIZE   2048   // I + H concatenated reduction dim
#define KSLICE  512    // K per split-K slice (4 slices)
#define NSLICES 4

typedef __attribute__((ext_vector_type(16))) __bf16 bf16x16;
typedef __attribute__((ext_vector_type(8)))  float  f32x8;

union FragB { bf16x16 v; uint4 q[2]; };

__device__ __forceinline__ unsigned int f2bf(float f) {
  // round-to-nearest-even f32 -> bf16
  unsigned int u = __float_as_uint(f);
  return (u + 0x7FFFu + ((u >> 16) & 1u)) >> 16;
}
__device__ __forceinline__ unsigned int pack2(float lo, float hi) {
  return f2bf(lo) | (f2bf(hi) << 16);
}

// A fragment 16x32 bf16 from row-major bf16 (lane&15 = M row, lane>>4 = K half)
__device__ __forceinline__ FragB loadA_bf16(const unsigned short* p) {
  FragB a;
  a.q[0] = *(const uint4*)(p);        // K = +0..7   (of this lane's subrange)
  a.q[1] = *(const uint4*)(p + 16);   // K = +16..23
  return a;
}
// Same fragment but from f32 memory, converting to bf16 inline
__device__ __forceinline__ FragB loadA_f32(const float* p) {
  float4 x0 = *(const float4*)(p);
  float4 x1 = *(const float4*)(p + 4);
  float4 x2 = *(const float4*)(p + 16);
  float4 x3 = *(const float4*)(p + 20);
  FragB a;
  a.q[0] = make_uint4(pack2(x0.x, x0.y), pack2(x0.z, x0.w),
                      pack2(x1.x, x1.y), pack2(x1.z, x1.w));
  a.q[1] = make_uint4(pack2(x2.x, x2.y), pack2(x2.z, x2.w),
                      pack2(x3.x, x3.y), pack2(x3.z, x3.w));
  return a;
}

// Load 4 B fragments (32x16 each) then issue 4 WMMAs — loads batched first so
// the scheduler can overlap them instead of wait(0) before every WMMA.
__device__ __forceinline__ void mma4(const FragB a, const unsigned short* __restrict__ Wc,
                                     int nb, int l15, int kglob, f32x8 acc[4]) {
  const unsigned short* w0 = Wc + (size_t)(nb + l15) * KSIZE + kglob;
  const unsigned short* w1 = w0 + (size_t)16 * KSIZE;
  const unsigned short* w2 = w0 + (size_t)32 * KSIZE;
  const unsigned short* w3 = w0 + (size_t)48 * KSIZE;
  FragB b0, b1, b2, b3;
  b0.q[0] = *(const uint4*)(w0); b0.q[1] = *(const uint4*)(w0 + 16);
  b1.q[0] = *(const uint4*)(w1); b1.q[1] = *(const uint4*)(w1 + 16);
  b2.q[0] = *(const uint4*)(w2); b2.q[1] = *(const uint4*)(w2 + 16);
  b3.q[0] = *(const uint4*)(w3); b3.q[1] = *(const uint4*)(w3 + 16);
  acc[0] = __builtin_amdgcn_wmma_f32_16x16x32_bf16(false, a.v, false, b0.v, (short)0, acc[0], false, false);
  acc[1] = __builtin_amdgcn_wmma_f32_16x16x32_bf16(false, a.v, false, b1.v, (short)0, acc[1], false, false);
  acc[2] = __builtin_amdgcn_wmma_f32_16x16x32_bf16(false, a.v, false, b2.v, (short)0, acc[2], false, false);
  acc[3] = __builtin_amdgcn_wmma_f32_16x16x32_bf16(false, a.v, false, b3.v, (short)0, acc[3], false, false);
}

// ---------------------------------------------------------------------------
// Prep: Wc[g][k] = (k<I ? w_ih[g][k] : w_hh[g][k-I]) as bf16 ; bias = b_ih+b_hh
// ---------------------------------------------------------------------------
__global__ __launch_bounds__(256) void prep_weights(
    const float* __restrict__ w_ih, const float* __restrict__ w_hh,
    const float* __restrict__ b_ih, const float* __restrict__ b_hh,
    unsigned short* __restrict__ Wc, float* __restrict__ biasc)
{
  const long long idx = (long long)blockIdx.x * blockDim.x + threadIdx.x; // < 4096*2048
  const int g = (int)(idx >> 11);
  const int k = (int)(idx & 2047);
  const float v = (k < ISIZE) ? w_ih[(size_t)g * ISIZE + k]
                              : w_hh[(size_t)g * HSIZE + (k - ISIZE)];
  Wc[idx] = (unsigned short)f2bf(v);
  if (idx < GSIZE) biasc[idx] = b_ih[idx] + b_hh[idx];
}

// hstate (bf16) and cstate (f32) for both layers: 2*B*H elements
__global__ __launch_bounds__(256) void prep_state(
    const float* __restrict__ hx, const float* __restrict__ cx,
    unsigned short* __restrict__ hstate, float* __restrict__ cstate)
{
  const int idx = blockIdx.x * blockDim.x + threadIdx.x; // < 2*B*H
  hstate[idx] = (unsigned short)f2bf(hx[idx]);
  cstate[idx] = cx[idx];
}

// ---------------------------------------------------------------------------
// One recurrent-step GEMM with split-K x4:
//   gatesP[slice][64][4096] = partial of [x_t | h] @ Wc^T over K slice
// grid = (32, 4) x 256 threads (8 waves). Wave w: M-strip (w&3)*16,
// N-strip blockIdx.x*128 + (w>>2)*64. blockIdx.y = K slice of 512.
// Slices 0,1 cover x (f32 for layer 0, bf16 for layer 1); slices 2,3 cover h.
// Fragments stream straight from L2 via 128-bit loads in ISA-native layout.
// ---------------------------------------------------------------------------
template<bool XF32>
__global__ __launch_bounds__(256) void lstm_gemm_step(
    const void* __restrict__ xt,               // [64,1024] f32 (XF32) or bf16
    const unsigned short* __restrict__ hstate, // [64,1024] bf16
    const unsigned short* __restrict__ Wc,     // [4096,2048] bf16
    float* __restrict__ gatesP)                // [4][64][4096] f32 partials
{
  const int lane  = threadIdx.x & 31;
  const int wave  = threadIdx.x >> 5;
  const int l15   = lane & 15;
  const int lhalf = lane >> 4;               // 0 or 1
  const int m0    = (wave & 3) * 16;
  const int nb    = blockIdx.x * 128 + (wave >> 2) * 64;
  const int arow  = m0 + l15;                // batch row 0..63
  const int ks    = blockIdx.y;              // 0..3
  const int kbase = ks * KSLICE;             // global K start of this slice

  f32x8 acc[4] = {};

  if (ks < 2) {
    // ---- x part of the reduction (branch-free inner loop)
    if (XF32) {
      const float* xrow = (const float*)xt + (size_t)arow * ISIZE + kbase + lhalf * 8;
#pragma unroll 4
      for (int k = 0; k < KSLICE; k += 32) {
        FragB a = loadA_f32(xrow + k);
        mma4(a, Wc, nb, l15, kbase + k + lhalf * 8, acc);
      }
    } else {
      const unsigned short* xrow = (const unsigned short*)xt + (size_t)arow * ISIZE + kbase + lhalf * 8;
#pragma unroll 4
      for (int k = 0; k < KSLICE; k += 32) {
        FragB a = loadA_bf16(xrow + k);
        mma4(a, Wc, nb, l15, kbase + k + lhalf * 8, acc);
      }
    }
  } else {
    // ---- h part of the reduction
    const int hb = kbase - ISIZE;            // 0 or 512 within hstate
    const unsigned short* hrow = hstate + (size_t)arow * HSIZE + hb + lhalf * 8;
#pragma unroll 4
    for (int k = 0; k < KSLICE; k += 32) {
      FragB a = loadA_bf16(hrow + k);
      mma4(a, Wc, nb, l15, kbase + k + lhalf * 8, acc);
    }
  }

  // ---- Store D (16x16 f32 layout: lane&15 = N col, (lane>>4)*8 + vgpr = M row)
  float* gbase = gatesP + (size_t)ks * BATCH * GSIZE;
#pragma unroll
  for (int j = 0; j < 4; ++j) {
    const int n = nb + 16 * j + l15;
    float* gp = gbase + (size_t)(m0 + lhalf * 8) * GSIZE + n;
#pragma unroll
    for (int v = 0; v < 8; ++v)
      gp[(size_t)v * GSIZE] = acc[j][v];
  }
}

// ---------------------------------------------------------------------------
// Pointwise LSTM gating: sums the 4 split-K partials + bias, applies gates.
// ---------------------------------------------------------------------------
__global__ __launch_bounds__(256) void lstm_pointwise(
    const float* __restrict__ gatesP,         // [4][64][4096]
    const float* __restrict__ biasc,          // [4096]
    float* __restrict__ cstate,               // [64,1024] in/out
    unsigned short* __restrict__ hstate,      // [64,1024] bf16 out
    float* __restrict__ out_f32,              // nullable [64,1024]
    unsigned short* __restrict__ out_bf16,    // nullable [64,1024]
    float* __restrict__ hn,                   // nullable (last step)
    float* __restrict__ cn)                   // nullable (last step)
{
  const int idx = blockIdx.x * blockDim.x + threadIdx.x; // < 64*1024
  const int b = idx >> 10;
  const int h = idx & 1023;
  const size_t S = (size_t)BATCH * GSIZE;
  const float* g = gatesP + (size_t)b * GSIZE + h;
  const float ig = biasc[h]             + g[0]         + g[S]             + g[2*S]             + g[3*S];
  const float fg = biasc[h + HSIZE]     + g[HSIZE]     + g[S + HSIZE]     + g[2*S + HSIZE]     + g[3*S + HSIZE];
  const float gg = biasc[h + 2 * HSIZE] + g[2 * HSIZE] + g[S + 2 * HSIZE] + g[2*S + 2 * HSIZE] + g[3*S + 2 * HSIZE];
  const float og = biasc[h + 3 * HSIZE] + g[3 * HSIZE] + g[S + 3 * HSIZE] + g[2*S + 3 * HSIZE] + g[3*S + 3 * HSIZE];
  const float si = 1.0f / (1.0f + __expf(-ig));
  const float sf = 1.0f / (1.0f + __expf(-fg));
  const float so = 1.0f / (1.0f + __expf(-og));
  const float c  = sf * cstate[idx] + si * tanhf(gg);
  const float hh = so * tanhf(c);
  cstate[idx] = c;
  hstate[idx] = (unsigned short)f2bf(hh);
  if (out_f32)  out_f32[idx]  = hh;
  if (out_bf16) out_bf16[idx] = (unsigned short)f2bf(hh);
  if (hn) hn[idx] = hh;
  if (cn) cn[idx] = c;
}

// ---------------------------------------------------------------------------
extern "C" void kernel_launch(void* const* d_in, const int* in_sizes, int n_in,
                              void* d_out, int out_size, void* d_ws, size_t ws_size,
                              hipStream_t stream)
{
  (void)in_sizes; (void)n_in; (void)out_size; (void)ws_size;

  const float* x     = (const float*)d_in[0];
  const float* hx    = (const float*)d_in[1];
  const float* cx    = (const float*)d_in[2];
  const float* w_ih0 = (const float*)d_in[3];
  const float* w_hh0 = (const float*)d_in[4];
  const float* b_ih0 = (const float*)d_in[5];
  const float* b_hh0 = (const float*)d_in[6];
  const float* w_ih1 = (const float*)d_in[7];
  const float* w_hh1 = (const float*)d_in[8];
  const float* b_ih1 = (const float*)d_in[9];
  const float* b_hh1 = (const float*)d_in[10];

  char* p = (char*)d_ws;
  auto carve = [&](size_t bytes) {
    char* r = p;
    p += (bytes + 255) & ~(size_t)255;
    return r;
  };

  unsigned short* Wc0   = (unsigned short*)carve((size_t)GSIZE * KSIZE * 2);
  unsigned short* Wc1   = (unsigned short*)carve((size_t)GSIZE * KSIZE * 2);
  float*          bias0 = (float*)carve((size_t)GSIZE * 4);
  float*          bias1 = (float*)carve((size_t)GSIZE * 4);
  unsigned short* Xb1   = (unsigned short*)carve((size_t)T_STEPS * BATCH * HSIZE * 2);
  unsigned short* hs    = (unsigned short*)carve((size_t)2 * BATCH * HSIZE * 2);
  float*          cs    = (float*)carve((size_t)2 * BATCH * HSIZE * 4);
  float*          gatesP= (float*)carve((size_t)NSLICES * BATCH * GSIZE * 4);

  unsigned short* hs0 = hs;
  unsigned short* hs1 = hs + (size_t)BATCH * HSIZE;
  float* cs0 = cs;
  float* cs1 = cs + (size_t)BATCH * HSIZE;

  prep_weights<<<(GSIZE * KSIZE) / 256, 256, 0, stream>>>(w_ih0, w_hh0, b_ih0, b_hh0, Wc0, bias0);
  prep_weights<<<(GSIZE * KSIZE) / 256, 256, 0, stream>>>(w_ih1, w_hh1, b_ih1, b_hh1, Wc1, bias1);
  prep_state<<<(2 * BATCH * HSIZE) / 256, 256, 0, stream>>>(hx, cx, hs, cs);

  float* out = (float*)d_out;                                   // [T,B,H]
  float* hn  = out + (size_t)T_STEPS * BATCH * HSIZE;           // [2,B,H]
  float* cn  = hn + (size_t)2 * BATCH * HSIZE;                  // [2,B,H]

  const dim3 ggrid(32, NSLICES);

  // ---- Layer 0: f32 x input, bf16 output feeds layer 1
  for (int t = 0; t < T_STEPS; ++t) {
    lstm_gemm_step<true><<<ggrid, 256, 0, stream>>>(
        x + (size_t)t * BATCH * ISIZE, hs0, Wc0, gatesP);
    const bool last = (t == T_STEPS - 1);
    lstm_pointwise<<<(BATCH * HSIZE) / 256, 256, 0, stream>>>(
        gatesP, bias0, cs0, hs0,
        nullptr, Xb1 + (size_t)t * BATCH * HSIZE,
        last ? hn : nullptr, last ? cn : nullptr);
  }

  // ---- Layer 1: bf16 input, f32 output to d_out
  for (int t = 0; t < T_STEPS; ++t) {
    lstm_gemm_step<false><<<ggrid, 256, 0, stream>>>(
        Xb1 + (size_t)t * BATCH * HSIZE, hs1, Wc1, gatesP);
    const bool last = (t == T_STEPS - 1);
    lstm_pointwise<<<(BATCH * HSIZE) / 256, 256, 0, stream>>>(
        gatesP, bias1, cs1, hs1,
        out + (size_t)t * BATCH * HSIZE, nullptr,
        last ? hn + BATCH * HSIZE : nullptr, last ? cn + BATCH * HSIZE : nullptr);
  }
}
